// DynamicSparseAttention_5772436045828
// MI455X (gfx1250) — matrix operand order repaired and checked by reading, not compiled, not run on hardware
//
#include <hip/hip_runtime.h>
#include <hip/hip_bf16.h>
#include <stdint.h>
#include <stddef.h>

// ---------------- constants (match reference) ----------------
#define BN 2
#define SN 2048
#define DN 512
#define NHEAD 8
#define HDIM 64
#define NROWS (BN * SN)            // 4096
#define MASK_WORDS (SN / 32)       // 64 words per (b,q) row

typedef _Float16 v16h __attribute__((ext_vector_type(16)));
typedef float v8f __attribute__((ext_vector_type(8)));

union U8f  { v8f v;  float f[8];      };
union U16h { v16h v; _Float16 h[16];  };

struct BStats {
    float thr[BN], lo[BN], hi[BN], comp[BN];
    int has_kp[BN];
    unsigned dens[BN];
    int needed[BN];
};

__device__ __forceinline__ void zero8(U8f& u) {
#pragma unroll
    for (int i = 0; i < 8; ++i) u.f[i] = 0.f;
}

// ---- CDNA5 async Global->LDS copy (ASYNCcnt path), inline asm (toolchain-portable) ----
// LDS dest VGPR carries the LDS byte offset (low 32 bits of a generic shared pointer are
// the LDS offset per the aperture mapping); global address in a 64-bit VGPR pair.
__device__ __forceinline__ void async_g2l_b128(void* lds_ptr, const void* gptr) {
    unsigned loff = (unsigned)(uintptr_t)lds_ptr;
    asm volatile("global_load_async_to_lds_b128 %0, %1, off"
                 :: "v"(loff), "v"(gptr) : "memory");
}
__device__ __forceinline__ void wait_async0() {
    asm volatile("s_wait_asynccnt 0x0" ::: "memory");
}

// ================= LayerNorm (f32 in, f32+f16 out) =================
__global__ void k_layernorm(const float* __restrict__ x,
                            const float* __restrict__ gam,
                            const float* __restrict__ bet,
                            float* __restrict__ o32,
                            _Float16* __restrict__ o16) {
    __shared__ float sm[256];
    const int t = threadIdx.x;
    const size_t row = blockIdx.x;
    const float* xr = x + row * DN;
    float a = xr[t], b = xr[t + 256];

    sm[t] = a + b; __syncthreads();
    for (int s = 128; s > 0; s >>= 1) { if (t < s) sm[t] += sm[t + s]; __syncthreads(); }
    const float mean = sm[0] * (1.f / DN);
    __syncthreads();
    float ca = a - mean, cb = b - mean;
    sm[t] = ca * ca + cb * cb; __syncthreads();
    for (int s = 128; s > 0; s >>= 1) { if (t < s) sm[t] += sm[t + s]; __syncthreads(); }
    const float var = sm[0] * (1.f / DN);          // ddof=0, matches jnp.var
    const float rstd = rsqrtf(var + 1e-5f);

    float y0 = ca * rstd * gam[t] + bet[t];
    float y1 = cb * rstd * gam[t + 256] + bet[t + 256];
    o16[row * DN + t]       = (_Float16)y0;
    o16[row * DN + t + 256] = (_Float16)y1;
    if (o32) { o32[row * DN + t] = y0; o32[row * DN + t + 256] = y1; }
}

// ============ weight transpose + f16 convert: Wt[d][o] = W[o][d] ============
__global__ void k_cvt_w(const float* __restrict__ W, _Float16* __restrict__ Wt) {
    int idx = blockIdx.x * 256 + threadIdx.x;      // 512*512
    int o = idx >> 9, d = idx & 511;
    Wt[(size_t)d * DN + o] = (_Float16)W[(size_t)o * DN + d];
}

// ============ per-query local window via tiny MLP ============
__global__ void k_win(const float* __restrict__ qn, const float* __restrict__ W1,
                      const float* __restrict__ b1, const float* __restrict__ W2,
                      const float* __restrict__ b2, int* __restrict__ win) {
    __shared__ float sm[128];
    const int t = threadIdx.x;                     // 128 hidden units
    const float* row = qn + (size_t)blockIdx.x * DN;
    const float* wr = W1 + (size_t)t * DN;
    float acc = b1[t];
    for (int d = 0; d < DN; ++d) acc += row[d] * wr[d];
    sm[t] = fmaxf(acc, 0.f) * W2[t];
    __syncthreads();
    for (int s = 64; s > 0; s >>= 1) { if (t < s) sm[t] += sm[t + s]; __syncthreads(); }
    if (t == 0) {
        float z = sm[0] + b2[0];
        float sg = 1.f / (1.f + __expf(-z));
        float wv = rintf(64.f * (0.5f + sg));      // round-half-even like jnp.round
        wv = fminf(fmaxf(wv, 1.f), 128.f);         // MAX_WIN = 128
        win[blockIdx.x] = (int)wv;
    }
}

// ============ per-row stats: dm (multi-scale change), mag, ch, var(ddof=1) ============
__global__ void k_rowstats(const float* __restrict__ qn, float* __restrict__ dm,
                           float* __restrict__ mag, float* __restrict__ ch,
                           float* __restrict__ varr) {
    __shared__ float sm[128];
    const int t = threadIdx.x;
    const int bi = blockIdx.x;
    const int s = bi & (SN - 1);
    const float* row = qn + (size_t)bi * DN;
    const bool v1 = (s + 1) < SN, v2 = (s + 2) < SN, v3 = (s + 3) < SN, v5 = (s + 5) < SN;

    float p1 = 0, p2 = 0, d1 = 0, d2 = 0, d3 = 0, d5 = 0;
    for (int d = t; d < DN; d += 128) {
        float x = row[d];
        p1 += x; p2 += x * x;
        if (v1) d1 += fabsf(row[1 * DN + d] - x);
        if (v2) d2 += fabsf(row[2 * DN + d] - x);
        if (v3) d3 += fabsf(row[3 * DN + d] - x);
        if (v5) d5 += fabsf(row[5 * DN + d] - x);
    }
    auto red = [&](float v) -> float {
        sm[t] = v; __syncthreads();
        for (int st = 64; st > 0; st >>= 1) { if (t < st) sm[t] += sm[t + st]; __syncthreads(); }
        float r = sm[0]; __syncthreads(); return r;
    };
    float S1 = red(p1), S2 = red(p2), D1 = red(d1), D2 = red(d2), D3 = red(d3), D5 = red(d5);
    if (t == 0) {
        const float inv = 1.f / DN;
        float dmv = 0.f;
        if (v1) dmv += 0.4f * (D1 * inv);
        if (v2) dmv += 0.3f * (D2 * inv) * 0.5f;
        if (v3) dmv += 0.2f * (D3 * inv) * (1.f / 3.f);
        if (v5) dmv += 0.1f * (D5 * inv) * 0.2f;
        dm[bi]   = dmv;
        mag[bi]  = sqrtf(S2);
        ch[bi]   = v1 ? D1 * inv : 0.f;
        varr[bi] = (S2 - S1 * S1 * inv) * (1.f / (DN - 1));
    }
}

// ============ per-batch stats ============
__global__ void k_bstats(const float* __restrict__ dm, const float* __restrict__ mag,
                         const float* __restrict__ ch, const float* __restrict__ varr,
                         BStats* __restrict__ bs) {
    __shared__ float sm[256];
    const int t = threadIdx.x;
    const int b = blockIdx.x;
    const size_t base = (size_t)b * SN;
    float s1 = 0, s2 = 0, sv = 0, lo = 3e38f, hi = -3e38f;
    for (int i = t; i < SN; i += 256) {
        float d = dm[base + i];
        s1 += d; s2 += d * d; sv += varr[base + i];
        float imp = 0.5f * mag[base + i] + 0.5f * ch[base + i];
        lo = fminf(lo, imp); hi = fmaxf(hi, imp);
    }
    auto redS = [&](float v) -> float {
        sm[t] = v; __syncthreads();
        for (int st = 128; st > 0; st >>= 1) { if (t < st) sm[t] += sm[t + st]; __syncthreads(); }
        float r = sm[0]; __syncthreads(); return r;
    };
    auto redMin = [&](float v) -> float {
        sm[t] = v; __syncthreads();
        for (int st = 128; st > 0; st >>= 1) { if (t < st) sm[t] = fminf(sm[t], sm[t + st]); __syncthreads(); }
        float r = sm[0]; __syncthreads(); return r;
    };
    auto redMax = [&](float v) -> float {
        sm[t] = v; __syncthreads();
        for (int st = 128; st > 0; st >>= 1) { if (t < st) sm[t] = fmaxf(sm[t], sm[t + st]); __syncthreads(); }
        float r = sm[0]; __syncthreads(); return r;
    };
    float S1 = redS(s1), S2 = redS(s2), SV = redS(sv), LO = redMin(lo), HI = redMax(hi);
    if (t == 0) {
        float mean = S1 / SN;
        float sd = sqrtf(fmaxf((S2 - S1 * S1 / SN) / (SN - 1), 0.f));  // ddof=1
        bs->thr[b] = mean + 0.5f * sd;   // THRESHOLD = 0.5
        bs->lo[b] = LO; bs->hi[b] = HI;
        bs->comp[b] = SV / SN;
        bs->has_kp[b] = 0;
        bs->dens[b] = 0u;
    }
}

// ============ keypoint peaks ============
__global__ void k_kp(const float* __restrict__ dm, BStats* __restrict__ bs,
                     unsigned char* __restrict__ kp) {
    int idx = blockIdx.x * 256 + threadIdx.x;      // BN*SN
    int b = idx >> 11, s = idx & (SN - 1);
    float c = dm[idx];
    float l = (s > 0) ? dm[idx - 1] : -1.f;
    float r = (s < SN - 1) ? dm[idx + 1] : -1.f;
    float thr = bs->thr[b];
    bool peak = (c > l) && (c > r);
    bool k = (c > thr) && peak;
    if (s == 0 || s == SN - 1) k = k || (c > thr);
    kp[idx] = k ? (unsigned char)1 : (unsigned char)0;
    if (k) atomicOr(&bs->has_kp[b], 1);
}

// ============ stratified top-k global columns (deterministic top-4 per segment) ============
__global__ void k_topk(const float* __restrict__ mag, const float* __restrict__ ch,
                       unsigned char* __restrict__ colm) {
    if (threadIdx.x != 0) return;
    int b = blockIdx.x >> 2, layer = blockIdx.x & 3;
    size_t base = (size_t)b * SN + layer * 512;
    int chosen[4] = {-1, -1, -1, -1};
    for (int tsel = 0; tsel < 4; ++tsel) {
        float best = -3e38f; int bi = 0;
        for (int i = 0; i < 512; ++i) {
            bool used = false;
            for (int j = 0; j < tsel; ++j) if (chosen[j] == i) used = true;
            if (used) continue;
            float v = 0.5f * mag[base + i] + 0.5f * ch[base + i];
            if (v > best) { best = v; bi = i; }
        }
        chosen[tsel] = bi;
        colm[base + bi] = 1;
    }
}

// ============ assemble mask bitfield + density ============
__global__ void k_maskbuild(const int* __restrict__ win, const unsigned char* __restrict__ kp,
                            const unsigned char* __restrict__ colm, BStats* __restrict__ bs,
                            unsigned* __restrict__ mb) {
    int idx = blockIdx.x * 256 + threadIdx.x;      // BN*SN*64
    int b = idx >> 17;
    int rem = idx & (SN * MASK_WORDS - 1);
    int q = rem >> 6, w = rem & (MASK_WORDS - 1);
    int wq = win[(b << 11) + q];
    int hk = bs->has_kp[b];
    int kq = kp[(b << 11) + q];
    bool c5q = (q == 0) || (q == 511) || (q == 1023) || (q == 1535) || (q == 2047);
    unsigned word = 0;
    for (int bit = 0; bit < 32; ++bit) {
        int k = (w << 5) + bit;
        bool loc = (k >= q - wq) && (k <= q + wq);
        bool c5k = (k == 0) || (k == 511) || (k == 1023) || (k == 1535) || (k == 2047);
        bool kpt = hk ? ((kq != 0) || (kp[(b << 11) + k] != 0)) : (c5q || c5k);
        bool gl = colm[(b << 11) + k] != 0;
        if (loc || kpt || gl) word |= (1u << bit);
    }
    mb[idx] = word;
    atomicAdd(&bs->dens[b], (unsigned)__popc(word));
}

// ============ needed random fill count ============
__global__ void k_needed(BStats* __restrict__ bs) {
    if (threadIdx.x || blockIdx.x) return;
    float c0 = bs->comp[0], c1 = bs->comp[1];
    float cmin = fminf(c0, c1), cmax = fmaxf(c0, c1);
    for (int b = 0; b < BN; ++b) {
        float nc = (bs->comp[b] - cmin) / (cmax - cmin + 1e-6f);
        float ratio = fminf(fmaxf(0.3f * (0.5f + nc), 0.1f), 0.5f);
        float dens = (float)bs->dens[b] / (float)((size_t)SN * SN);
        float need = ceilf((ratio - dens) * (float)((size_t)SN * SN));
        bs->needed[b] = need > 0.f ? (int)need : 0;
    }
}

// ============ deterministic random fill (splitmix64) ============
__global__ void k_randfill(const BStats* __restrict__ bs, unsigned* __restrict__ mb) {
    int t = blockIdx.x * 256 + threadIdx.x;        // 65536 threads
    for (int b = 0; b < BN; ++b) {
        int nb = bs->needed[b];
        for (int i = t; i < nb; i += 65536) {
            uint64_t z = (((uint64_t)i << 1) | (unsigned)b) + 0x9E3779B97F4A7C15ull;
            z ^= z >> 30; z *= 0xBF58476D1CE4E5B9ull;
            z ^= z >> 27; z *= 0x94D049BB133111EBull;
            z ^= z >> 31;
            int q = (int)(z & (SN - 1));
            int k = (int)((z >> 11) & (SN - 1));
            atomicOr(&mb[((size_t)(b << 11) + q) * MASK_WORDS + (k >> 5)], 1u << (k & 31));
        }
    }
}

// ============ guarantee each row has >=1 set bit ============
__global__ void k_rowfix(unsigned* __restrict__ mb) {
    int idx = blockIdx.x * 256 + threadIdx.x;      // BN*SN
    size_t base = (size_t)idx * MASK_WORDS;
    unsigned acc = 0;
    for (int w = 0; w < MASK_WORDS; ++w) acc |= mb[base + w];
    if (!acc) {
        int q = idx & (SN - 1);
        mb[base + (q >> 5)] |= 1u << (q & 31);
    }
}

// ============ WMMA projection: out = A[4096,512] @ Wt + bias, f16 out ============
// mode 0: [B,H,S,64] layout (Q with scale=1/8, V with scale=1) ; mode 1: Kt [B,H,64,S]
__global__ __launch_bounds__(256) void k_proj(const _Float16* __restrict__ A,
                                              const _Float16* __restrict__ Wt,
                                              const float* __restrict__ bias,
                                              _Float16* __restrict__ out,
                                              int mode, float scale) {
    const int wave = blockIdx.x * 8 + (threadIdx.x >> 5);
    const int lane = threadIdx.x & 31, lo = lane & 15, g = lane >> 4;
    const int mt = wave >> 5, nt = wave & 31;      // 256 x 32 tiles
    const int m0 = mt * 16, n0 = nt * 16;

    U8f c; zero8(c);
    const _Float16* arowBase = A + (size_t)(m0 + lo) * DN;
    for (int d0 = 0; d0 < DN; d0 += 32) {
        U16h a, bf;
        const _Float16* ar = arowBase + d0;
        __builtin_prefetch(ar + 32, 0, 3);
#pragma unroll
        for (int h = 0; h < 8; ++h) { a.h[h] = ar[8 * g + h]; a.h[h + 8] = ar[16 + 8 * g + h]; }
        const _Float16* br = Wt + (size_t)d0 * DN + n0 + lo;
#pragma unroll
        for (int h = 0; h < 8; ++h) { bf.h[h] = br[(size_t)(8 * g + h) * DN]; bf.h[h + 8] = br[(size_t)(16 + 8 * g + h) * DN]; }
        c.v = __builtin_amdgcn_wmma_f32_16x16x32_f16(false, a.v, false, bf.v, (short)0, c.v,
                                                     false, false);
    }
    const int n = n0 + lo;
    const int head = n >> 6, dd = n & 63;
    const float bv = bias[n];
#pragma unroll
    for (int r = 0; r < 8; ++r) {
        int m = m0 + r + 8 * g;
        int b = m >> 11, s = m & (SN - 1);
        float val = (c.f[r] + bv) * scale;
        size_t oidx;
        if (mode == 1) oidx = ((size_t)(b * NHEAD + head) * HDIM + dd) * SN + s;
        else           oidx = ((size_t)(b * NHEAD + head) * SN + s) * HDIM + dd;
        out[oidx] = (_Float16)val;
    }
}

// ============ flash-style masked attention, one 16-row q tile per wave ============
// All 8 waves of a block share the same (b,head): K/V tiles are staged once per block
// into LDS via async Global->LDS (ASYNCcnt) and consumed from LDS by every wave.
__global__ __launch_bounds__(256) void k_attn(const _Float16* __restrict__ Qh,
                                              const _Float16* __restrict__ Kt,
                                              const _Float16* __restrict__ Vh,
                                              const unsigned* __restrict__ mb,
                                              _Float16* __restrict__ attn) {
    __shared__ _Float16 pbuf[8][512];              // per-wave 16x32 P tile
    __shared__ _Float16 ldsK[64 * 32];             // [d][k_local]  (4 KB)
    __shared__ _Float16 ldsV[32 * 64];             // [k_local][d]  (4 KB)
    const int t = threadIdx.x;
    const int w = t >> 5, lane = t & 31, lo = lane & 15, g = lane >> 4;
    const int gw = blockIdx.x * 8 + w;
    const int bh = gw >> 7, qt = gw & 127, b = bh >> 3, q0 = qt * 16;

    U16h qa0, qa1;
    const _Float16* qr = Qh + ((size_t)bh * SN + q0 + lo) * HDIM;
#pragma unroll
    for (int h = 0; h < 8; ++h) {
        qa0.h[h]     = qr[8 * g + h];       qa0.h[h + 8] = qr[16 + 8 * g + h];
        qa1.h[h]     = qr[32 + 8 * g + h];  qa1.h[h + 8] = qr[48 + 8 * g + h];
    }
    float mr[8], lr[8];
    U8f oa[4];
#pragma unroll
    for (int r = 0; r < 8; ++r) { mr[r] = -3.0e38f; lr[r] = 0.f; }
#pragma unroll
    for (int n = 0; n < 4; ++n) zero8(oa[n]);

    const unsigned* mbB = mb + (size_t)b * SN * MASK_WORDS;

    // per-thread staging assignments (256 threads cover 256 b128 chunks per tile)
    const int kd = t >> 2, kc = t & 3;             // K: d-row, 16B chunk
    const int vr = t >> 3, vc = t & 7;             // V: key-row, 16B chunk

    for (int kb = 0; kb < SN; kb += 32) {
        // ---- async stage K (64x32) and V (32x64) slabs into LDS ----
        async_g2l_b128(&ldsK[kd * 32 + kc * 8],
                       Kt + ((size_t)bh * HDIM + kd) * SN + kb + kc * 8);
        async_g2l_b128(&ldsV[vr * 64 + vc * 8],
                       Vh + ((size_t)bh * SN + kb + vr) * HDIM + vc * 8);
        wait_async0();
        __syncthreads();

        U8f s0, s1; zero8(s0); zero8(s1);
        {
            U16h bf;
            const _Float16* kp0 = &ldsK[lo];
#pragma unroll
            for (int h = 0; h < 8; ++h) { bf.h[h] = kp0[(8 * g + h) * 32]; bf.h[h + 8] = kp0[(16 + 8 * g + h) * 32]; }
            s0.v = __builtin_amdgcn_wmma_f32_16x16x32_f16(false, qa0.v, false, bf.v, (short)0, s0.v, false, false);
#pragma unroll
            for (int h = 0; h < 8; ++h) { bf.h[h] = kp0[(8 * g + h) * 32 + 16]; bf.h[h + 8] = kp0[(16 + 8 * g + h) * 32 + 16]; }
            s1.v = __builtin_amdgcn_wmma_f32_16x16x32_f16(false, qa0.v, false, bf.v, (short)0, s1.v, false, false);
            const _Float16* kp1 = &ldsK[32 * 32 + lo];
#pragma unroll
            for (int h = 0; h < 8; ++h) { bf.h[h] = kp1[(8 * g + h) * 32]; bf.h[h + 8] = kp1[(16 + 8 * g + h) * 32]; }
            s0.v = __builtin_amdgcn_wmma_f32_16x16x32_f16(false, qa1.v, false, bf.v, (short)0, s0.v, false, false);
#pragma unroll
            for (int h = 0; h < 8; ++h) { bf.h[h] = kp1[(8 * g + h) * 32 + 16]; bf.h[h + 8] = kp1[(16 + 8 * g + h) * 32 + 16]; }
            s1.v = __builtin_amdgcn_wmma_f32_16x16x32_f16(false, qa1.v, false, bf.v, (short)0, s1.v, false, false);
        }
        // mask (reference passes ~sparse: participate where mask bit == 0)
#pragma unroll
        for (int r = 0; r < 8; ++r) {
            int q = q0 + r + 8 * g;
            const unsigned* mw = mbB + (size_t)q * MASK_WORDS;
            int k0 = kb + lo, k1 = kb + 16 + lo;
            if ((mw[k0 >> 5] >> (k0 & 31)) & 1u) s0.f[r] = -3.0e38f;
            if ((mw[k1 >> 5] >> (k1 & 31)) & 1u) s1.f[r] = -3.0e38f;
        }
        float corr[8];
#pragma unroll
        for (int r = 0; r < 8; ++r) {
            float v = fmaxf(s0.f[r], s1.f[r]);
            v = fmaxf(v, __shfl_xor(v, 1, 32));
            v = fmaxf(v, __shfl_xor(v, 2, 32));
            v = fmaxf(v, __shfl_xor(v, 4, 32));
            v = fmaxf(v, __shfl_xor(v, 8, 32));
            float mn = fmaxf(mr[r], v);
            corr[r] = __expf(mr[r] - mn);
            mr[r] = mn;
            float p0 = __expf(s0.f[r] - mn), p1 = __expf(s1.f[r] - mn);
            pbuf[w][(r + 8 * g) * 32 + lo]      = (_Float16)p0;
            pbuf[w][(r + 8 * g) * 32 + 16 + lo] = (_Float16)p1;
            float ps = p0 + p1;
            ps += __shfl_xor(ps, 1, 32);
            ps += __shfl_xor(ps, 2, 32);
            ps += __shfl_xor(ps, 4, 32);
            ps += __shfl_xor(ps, 8, 32);
            lr[r] = lr[r] * corr[r] + ps;
        }
        // re-stripe P from C-layout (via LDS) into A-layout, then P @ V (from LDS)
        U16h pa;
        const _Float16* pr = &pbuf[w][lo * 32];
#pragma unroll
        for (int h = 0; h < 8; ++h) { pa.h[h] = pr[8 * g + h]; pa.h[h + 8] = pr[16 + 8 * g + h]; }
        const _Float16* vb = &ldsV[lo];
#pragma unroll
        for (int n = 0; n < 4; ++n) {
#pragma unroll
            for (int r = 0; r < 8; ++r) oa[n].f[r] *= corr[r];
            U16h bf;
#pragma unroll
            for (int h = 0; h < 8; ++h) { bf.h[h] = vb[(8 * g + h) * 64 + n * 16]; bf.h[h + 8] = vb[(16 + 8 * g + h) * 64 + n * 16]; }
            oa[n].v = __builtin_amdgcn_wmma_f32_16x16x32_f16(false, pa.v, false, bf.v, (short)0, oa[n].v, false, false);
        }
        __syncthreads();   // protect ldsK/ldsV before next iteration's async overwrite
    }
    const int head = bh & 7;
#pragma unroll
    for (int r = 0; r < 8; ++r) {
        float invl = 1.f / fmaxf(lr[r], 1e-30f);
        int q = q0 + r + 8 * g;
        size_t obase = ((size_t)b * SN + q) * DN + head * HDIM;
#pragma unroll
        for (int n = 0; n < 4; ++n)
            attn[obase + n * 16 + lo] = (_Float16)(oa[n].f[r] * invl);
    }
}

// ============ output projection + bias + residual, f32 out ============
__global__ __launch_bounds__(256) void k_outgemm(const _Float16* __restrict__ A,
                                                 const _Float16* __restrict__ Wt,
                                                 const float* __restrict__ bias,
                                                 const float* __restrict__ resid,
                                                 float* __restrict__ out) {
    const int wave = blockIdx.x * 8 + (threadIdx.x >> 5);
    const int lane = threadIdx.x & 31, lo = lane & 15, g = lane >> 4;
    const int mt = wave >> 5, nt = wave & 31;
    const int m0 = mt * 16, n0 = nt * 16;

    U8f c; zero8(c);
    const _Float16* arowBase = A + (size_t)(m0 + lo) * DN;
    for (int d0 = 0; d0 < DN; d0 += 32) {
        U16h a, bf;
        const _Float16* ar = arowBase + d0;
#pragma unroll
        for (int h = 0; h < 8; ++h) { a.h[h] = ar[8 * g + h]; a.h[h + 8] = ar[16 + 8 * g + h]; }
        const _Float16* br = Wt + (size_t)d0 * DN + n0 + lo;
#pragma unroll
        for (int h = 0; h < 8; ++h) { bf.h[h] = br[(size_t)(8 * g + h) * DN]; bf.h[h + 8] = br[(size_t)(16 + 8 * g + h) * DN]; }
        c.v = __builtin_amdgcn_wmma_f32_16x16x32_f16(false, a.v, false, bf.v, (short)0, c.v,
                                                     false, false);
    }
    const int n = n0 + lo;
    const float bv = bias[n];
#pragma unroll
    for (int r = 0; r < 8; ++r) {
        int m = m0 + r + 8 * g;
        out[(size_t)m * DN + n] = c.f[r] + bv + resid[(size_t)m * DN + n];
    }
}

// ================= workspace layout =================
static constexpr size_t SZ_ROWS_F32 = (size_t)NROWS * DN * 4;   // 8 MB
static constexpr size_t SZ_ROWS_F16 = (size_t)NROWS * DN * 2;   // 4 MB
static constexpr size_t SZ_W_F16    = (size_t)DN * DN * 2;      // 512 KB

static constexpr size_t OFF_QN32 = 0;
static constexpr size_t OFF_QN16 = OFF_QN32 + SZ_ROWS_F32;
static constexpr size_t OFF_KN16 = OFF_QN16 + SZ_ROWS_F16;
static constexpr size_t OFF_VN16 = OFF_KN16 + SZ_ROWS_F16;
static constexpr size_t OFF_WQT  = OFF_VN16 + SZ_ROWS_F16;
static constexpr size_t OFF_WKT  = OFF_WQT + SZ_W_F16;
static constexpr size_t OFF_WVT  = OFF_WKT + SZ_W_F16;
static constexpr size_t OFF_WOT  = OFF_WVT + SZ_W_F16;
static constexpr size_t OFF_QH   = OFF_WOT + SZ_W_F16;
static constexpr size_t OFF_KT   = OFF_QH + SZ_ROWS_F16;
static constexpr size_t OFF_VH   = OFF_KT + SZ_ROWS_F16;
static constexpr size_t OFF_ATT  = OFF_VH + SZ_ROWS_F16;
static constexpr size_t OFF_MASK = OFF_ATT + SZ_ROWS_F16;
static constexpr size_t SZ_MASK  = (size_t)BN * SN * MASK_WORDS * 4;
static constexpr size_t OFF_WIN  = OFF_MASK + SZ_MASK;
static constexpr size_t OFF_DM   = OFF_WIN + (size_t)NROWS * 4;
static constexpr size_t OFF_MAG  = OFF_DM + (size_t)NROWS * 4;
static constexpr size_t OFF_CH   = OFF_MAG + (size_t)NROWS * 4;
static constexpr size_t OFF_VAR  = OFF_CH + (size_t)NROWS * 4;
static constexpr size_t OFF_KP   = OFF_VAR + (size_t)NROWS * 4;
static constexpr size_t OFF_COLM = OFF_KP + (size_t)NROWS;
static constexpr size_t OFF_BST  = OFF_COLM + (size_t)NROWS;

extern "C" void kernel_launch(void* const* d_in, const int* in_sizes, int n_in,
                              void* d_out, int out_size, void* d_ws, size_t ws_size,
                              hipStream_t stream) {
    (void)in_sizes; (void)n_in; (void)out_size; (void)ws_size;
    const float* q_in  = (const float*)d_in[0];
    const float* k_in  = (const float*)d_in[1];
    const float* v_in  = (const float*)d_in[2];
    const float* lqg   = (const float*)d_in[3];
    const float* lqb   = (const float*)d_in[4];
    const float* lkg   = (const float*)d_in[5];
    const float* lkb   = (const float*)d_in[6];
    const float* lvg   = (const float*)d_in[7];
    const float* lvb   = (const float*)d_in[8];
    const float* Wq    = (const float*)d_in[9];
    const float* bq    = (const float*)d_in[10];
    const float* Wk    = (const float*)d_in[11];
    const float* bk    = (const float*)d_in[12];
    const float* Wv    = (const float*)d_in[13];
    const float* bv    = (const float*)d_in[14];
    const float* Wo    = (const float*)d_in[15];
    const float* bo    = (const float*)d_in[16];
    const float* Ww1   = (const float*)d_in[17];
    const float* bw1   = (const float*)d_in[18];
    const float* Ww2   = (const float*)d_in[19];
    const float* bw2   = (const float*)d_in[20];

    char* w = (char*)d_ws;
    float*         qn32  = (float*)(w + OFF_QN32);
    _Float16*      qn16  = (_Float16*)(w + OFF_QN16);
    _Float16*      kn16  = (_Float16*)(w + OFF_KN16);
    _Float16*      vn16  = (_Float16*)(w + OFF_VN16);
    _Float16*      wqt   = (_Float16*)(w + OFF_WQT);
    _Float16*      wkt   = (_Float16*)(w + OFF_WKT);
    _Float16*      wvt   = (_Float16*)(w + OFF_WVT);
    _Float16*      wot   = (_Float16*)(w + OFF_WOT);
    _Float16*      Qh    = (_Float16*)(w + OFF_QH);
    _Float16*      Ktb   = (_Float16*)(w + OFF_KT);
    _Float16*      Vh    = (_Float16*)(w + OFF_VH);
    _Float16*      attn  = (_Float16*)(w + OFF_ATT);
    unsigned*      mbuf  = (unsigned*)(w + OFF_MASK);
    int*           win   = (int*)(w + OFF_WIN);
    float*         dm    = (float*)(w + OFF_DM);
    float*         mag   = (float*)(w + OFF_MAG);
    float*         ch    = (float*)(w + OFF_CH);
    float*         varr  = (float*)(w + OFF_VAR);
    unsigned char* kp    = (unsigned char*)(w + OFF_KP);
    unsigned char* colm  = (unsigned char*)(w + OFF_COLM);
    BStats*        bst   = (BStats*)(w + OFF_BST);

    // 1) LayerNorm (qn kept in f32 for the mask pipeline, f16 for WMMA)
    k_layernorm<<<NROWS, 256, 0, stream>>>(q_in, lqg, lqb, qn32, qn16);
    k_layernorm<<<NROWS, 256, 0, stream>>>(k_in, lkg, lkb, nullptr, kn16);
    k_layernorm<<<NROWS, 256, 0, stream>>>(v_in, lvg, lvb, nullptr, vn16);

    // 2) Weight transpose + f16
    k_cvt_w<<<1024, 256, 0, stream>>>(Wq, wqt);
    k_cvt_w<<<1024, 256, 0, stream>>>(Wk, wkt);
    k_cvt_w<<<1024, 256, 0, stream>>>(Wv, wvt);
    k_cvt_w<<<1024, 256, 0, stream>>>(Wo, wot);

    // 3) Mask pipeline
    k_win<<<NROWS, 128, 0, stream>>>(qn32, Ww1, bw1, Ww2, bw2, win);
    k_rowstats<<<NROWS, 128, 0, stream>>>(qn32, dm, mag, ch, varr);
    k_bstats<<<BN, 256, 0, stream>>>(dm, mag, ch, varr, bst);
    hipMemsetAsync(colm, 0, (size_t)NROWS, stream);
    k_kp<<<NROWS / 256, 256, 0, stream>>>(dm, bst, kp);
    k_topk<<<BN * 4, 32, 0, stream>>>(mag, ch, colm);
    k_maskbuild<<<(BN * SN * MASK_WORDS) / 256, 256, 0, stream>>>(win, kp, colm, bst, mbuf);
    k_needed<<<1, 32, 0, stream>>>(bst);
    k_randfill<<<256, 256, 0, stream>>>(bst, mbuf);
    k_rowfix<<<NROWS / 256, 256, 0, stream>>>(mbuf);

    // 4) WMMA projections (Q pre-scaled by 1/sqrt(64); K stored transposed)
    k_proj<<<1024, 256, 0, stream>>>(qn16, wqt, bq, Qh, 0, 0.125f);
    k_proj<<<1024, 256, 0, stream>>>(kn16, wkt, bk, Ktb, 1, 1.0f);
    k_proj<<<1024, 256, 0, stream>>>(vn16, wvt, bv, Vh, 0, 1.0f);

    // 5) Masked flash attention on WMMA (async K/V staging into LDS)
    k_attn<<<256, 256, 0, stream>>>(Qh, Ktb, Vh, mbuf, attn);

    // 6) Output projection + residual
    k_outgemm<<<1024, 256, 0, stream>>>(attn, wot, bo, q_in, (float*)d_out);
}
